// Readout_21440476742383
// MI455X (gfx1250) — matrix operand order repaired and checked by reading, not compiled, not run on hardware
//
#include <hip/hip_runtime.h>
#include <hip/hip_bf16.h>

typedef __attribute__((ext_vector_type(8)))  _Float16 v8h;
typedef __attribute__((ext_vector_type(16))) _Float16 v16h;
typedef __attribute__((ext_vector_type(8)))  float    v8f;

#define NODES   200000
#define GRAPHS  2048
#define HID     200
#define EMB     200
#define DIN     400     // HID + EMB
#define KPAD    416     // DIN padded to multiple of 32
#define H1      256
#define H2      64
#define NKS1    (KPAD / 32)   // 13
#define NKS2    (H1 / 32)     // 8

// LDS strides chosen so fragment loads are 16B-aligned / bank-spread
#define AK_STRIDE 40    // halves: 64 rows x 32 kk (80 B/row)
#define AK_BUF    (64 * AK_STRIDE)
#define F1_STRIDE 264   // halves: 64 rows x 256 col (528 B/row)
#define F2_STRIDE 68    // floats: 64 rows x 64 col  (272 B/row)

// ---------------------------------------------------------------------------
// Prep: zero pooled accumulator; transpose W1 -> f16 [256][416] (K zero-padded)
// and W2 -> f16 [64][256] so B fragments are contiguous 32B per lane.
// ---------------------------------------------------------------------------
__global__ __launch_bounds__(256) void prep_kernel(const float* __restrict__ W1,
                                                   const float* __restrict__ W2,
                                                   _Float16* __restrict__ W1T,
                                                   _Float16* __restrict__ W2T,
                                                   float* __restrict__ gfeat) {
  const int i = blockIdx.x * blockDim.x + threadIdx.x;
  if (i < GRAPHS * H2) gfeat[i] = 0.0f;
  if (i < H1 * KPAD) {
    const int col = i / KPAD;
    const int k   = i - col * KPAD;
    W1T[i] = (k < DIN) ? (_Float16)W1[k * H1 + col] : (_Float16)0.0f;
  }
  if (i < H2 * H1) {
    const int n = i >> 8;
    const int k = i & 255;
    W2T[n * H1 + k] = (_Float16)W2[k * H2 + n];
  }
}

// ---------------------------------------------------------------------------
// Main: fused concat -> L1(relu) -> L2(relu) -> segmented LDS reduce -> atomics
// 256 threads = 8 waves; M-tile = 64 nodes; wave -> (m = wave>>1, nh = wave&1).
// Double-buffered A tile: one barrier per k-step, staging overlaps WMMA.
// ---------------------------------------------------------------------------
__global__ __launch_bounds__(256) void mlp_pool_kernel(
    const float* __restrict__ h, const float* __restrict__ x,
    const int* __restrict__ batch,
    const float* __restrict__ b1, const float* __restrict__ b2,
    const _Float16* __restrict__ W1T, const _Float16* __restrict__ W2T,
    float* __restrict__ gfeat) {
  __shared__ _Float16 Ak[2 * AK_BUF];       // 10,240 B (double buffered)
  __shared__ _Float16 F1[64 * F1_STRIDE];   // 33,792 B
  __shared__ float    F2[64 * F2_STRIDE];   // 17,408 B
  __shared__ int      gIdx[64];

  const int tid   = threadIdx.x;
  const int lane  = tid & 31;
  const int wave  = tid >> 5;
  const int base  = blockIdx.x * 64;

  const int m     = wave >> 1;     // 0..3 (M subtile of 16 rows)
  const int nh    = wave & 1;      // 0..1 (N half)
  const int nbase = nh * 128;
  const int lhalf = lane >> 4;     // 0 | 1
  const int l15   = lane & 15;

  if (tid < 64) gIdx[tid] = batch[base + tid];

  // Stage one 64x32 A chunk (branchless: clamped index + pointer select)
  auto stageA = [&](int ksn) {
    const int k0n = ksn * 32;
    _Float16* dst = &Ak[(ksn & 1) * AK_BUF];
#pragma unroll 1
    for (int i = tid; i < 64 * 32; i += 256) {
      const int row  = i >> 5, kk = i & 31;
      const int f    = k0n + kk;
      const int node = base + row;
      const int fc   = (f < DIN) ? f : 0;
      const float* p = (fc < HID) ? (h + node * HID + fc)
                                  : (x + node * EMB + (fc - HID));
      float v = *p;
      if (f >= DIN) v = 0.0f;
      dst[row * AK_STRIDE + kk] = (_Float16)v;
    }
  };

  // ---------------- Layer 1: [64 x 416] x [416 x 256] ----------------
  v8f acc[8];
#pragma unroll
  for (int t = 0; t < 8; ++t)
#pragma unroll
    for (int r = 0; r < 8; ++r) acc[t][r] = 0.0f;

  // Per-lane base pointers; tile offsets become constant IOFFSETs
  const _Float16* bp1 = W1T + (nbase + l15) * KPAD + lhalf * 16;
  const int aoff = (m * 16 + l15) * AK_STRIDE + lhalf * 16;

  stageA(0);
#pragma unroll 1
  for (int ks = 0; ks < NKS1; ++ks) {
    const int k0 = ks * 32;
    __syncthreads();
    if (ks + 1 < NKS1) stageA(ks + 1);      // overlaps with WMMAs below

    // Prefetch next k-slice of W1T
    __builtin_prefetch(bp1 + k0 + 32 + lane * 8, 0, 0);

    // A fragment (contiguous k = 16*lhalf + s, matching B)
    const _Float16* ap = &Ak[(ks & 1) * AK_BUF + aoff];
    const v8h alo = *(const v8h*)ap;
    const v8h ahi = *(const v8h*)(ap + 8);
    const v16h afrag = __builtin_shufflevector(alo, ahi,
        0, 1, 2, 3, 4, 5, 6, 7, 8, 9, 10, 11, 12, 13, 14, 15);

    // Preload all 8 B fragments: base + constant 13312B*t offsets
    const _Float16* bk = bp1 + k0;
    v16h bf[8];
#pragma unroll
    for (int t = 0; t < 8; ++t)
      bf[t] = *(const v16h*)(bk + t * (16 * KPAD));
#pragma unroll
    for (int t = 0; t < 8; ++t)
      acc[t] = __builtin_amdgcn_wmma_f32_16x16x32_f16(
          false, afrag, false, bf[t], (short)0, acc[t], false, false);
  }

  // Epilogue L1: bias + relu -> f16 in LDS
#pragma unroll
  for (int t = 0; t < 8; ++t) {
    const int col = nbase + t * 16 + l15;
    const float bias = b1[col];
#pragma unroll
    for (int r = 0; r < 8; ++r) {
      float v = acc[t][r] + bias;
      v = v > 0.0f ? v : 0.0f;
      F1[(m * 16 + r + lhalf * 8) * F1_STRIDE + col] = (_Float16)v;
    }
  }
  __syncthreads();

  // ---------------- Layer 2: [64 x 256] x [256 x 64] ----------------
  v8f a2[2];
#pragma unroll
  for (int t = 0; t < 2; ++t)
#pragma unroll
    for (int r = 0; r < 8; ++r) a2[t][r] = 0.0f;

  const int n2base = nh * 32;
  const _Float16* bp2 = W2T + (n2base + l15) * H1 + lhalf * 16;
  const int a2off = (m * 16 + l15) * F1_STRIDE + lhalf * 16;

#pragma unroll 1
  for (int ks = 0; ks < NKS2; ++ks) {
    const int k0 = ks * 32;
    const _Float16* ap = &F1[a2off + k0];
    const v8h alo = *(const v8h*)ap;
    const v8h ahi = *(const v8h*)(ap + 8);
    const v16h afrag = __builtin_shufflevector(alo, ahi,
        0, 1, 2, 3, 4, 5, 6, 7, 8, 9, 10, 11, 12, 13, 14, 15);
    v16h bf2[2];
#pragma unroll
    for (int t = 0; t < 2; ++t)
      bf2[t] = *(const v16h*)(bp2 + k0 + t * (16 * H1));
#pragma unroll
    for (int t = 0; t < 2; ++t)
      a2[t] = __builtin_amdgcn_wmma_f32_16x16x32_f16(
          false, afrag, false, bf2[t], (short)0, a2[t], false, false);
  }

  // Epilogue L2: bias + relu -> F2 (f32 in LDS)
#pragma unroll
  for (int t = 0; t < 2; ++t) {
    const int col = n2base + t * 16 + l15;
    const float bias = b2[col];
#pragma unroll
    for (int r = 0; r < 8; ++r) {
      float v = a2[t][r] + bias;
      v = v > 0.0f ? v : 0.0f;
      F2[(m * 16 + r + lhalf * 8) * F2_STRIDE + col] = v;
    }
  }
  __syncthreads();

  // Segmented reduction over sorted batch ids: ~1 atomic per (col, 16-row run)
  {
    const int col = tid & 63;
    const int rg  = tid >> 6;          // 0..3, rows rg*16 .. rg*16+15
    float run = 0.0f;
    int cur = gIdx[rg * 16];
#pragma unroll 1
    for (int r = 0; r < 16; ++r) {
      const int row = rg * 16 + r;
      const int g = gIdx[row];
      const float v = F2[row * F2_STRIDE + col];
      if (g != cur) {
        unsafeAtomicAdd(&gfeat[cur * H2 + col], run);
        run = 0.0f;
        cur = g;
      }
      run += v;
    }
    unsafeAtomicAdd(&gfeat[cur * H2 + col], run);
  }
}

// ---------------------------------------------------------------------------
// Readout: sigmoid(gfeat @ Wout + bout)
// ---------------------------------------------------------------------------
__global__ __launch_bounds__(256) void readout_kernel(const float* __restrict__ gfeat,
                                                      const float* __restrict__ Wout,
                                                      const float* __restrict__ bout,
                                                      float* __restrict__ out) {
  const int g = blockIdx.x * blockDim.x + threadIdx.x;
  if (g >= GRAPHS) return;
  float s = bout[0];
#pragma unroll
  for (int i = 0; i < H2; ++i) s += gfeat[g * H2 + i] * Wout[i];
  out[g] = 1.0f / (1.0f + expf(-s));
}

// ---------------------------------------------------------------------------
extern "C" void kernel_launch(void* const* d_in, const int* in_sizes, int n_in,
                              void* d_out, int out_size, void* d_ws, size_t ws_size,
                              hipStream_t stream) {
  const float* h     = (const float*)d_in[0];
  const float* x     = (const float*)d_in[1];
  const int*   batch = (const int*)d_in[2];
  const float* W1    = (const float*)d_in[3];
  const float* b1    = (const float*)d_in[4];
  const float* W2    = (const float*)d_in[5];
  const float* b2    = (const float*)d_in[6];
  const float* Wout  = (const float*)d_in[7];
  const float* bout  = (const float*)d_in[8];
  float* out = (float*)d_out;

  char* ws = (char*)d_ws;
  float*    gfeat = (float*)ws;                                              // 512 KB
  _Float16* W1T   = (_Float16*)(ws + GRAPHS * H2 * sizeof(float));           // 208 KB
  _Float16* W2T   = (_Float16*)(ws + GRAPHS * H2 * sizeof(float)
                                   + H1 * KPAD * sizeof(_Float16));          // 32 KB

  prep_kernel<<<(GRAPHS * H2 + 255) / 256, 256, 0, stream>>>(W1, W2, W1T, W2T, gfeat);
  mlp_pool_kernel<<<NODES / 64, 256, 0, stream>>>(h, x, batch, b1, b2, W1T, W2T, gfeat);
  readout_kernel<<<(GRAPHS + 255) / 256, 256, 0, stream>>>(gfeat, Wout, bout, out);
}
